// DPCA1D_30477087932759
// MI455X (gfx1250) — matrix operand order repaired and checked by reading, not compiled, not run on hardware
//
#include <hip/hip_runtime.h>
#include <hip/hip_bf16.h>

typedef __attribute__((ext_vector_type(16))) _Float16 v16h;
typedef __attribute__((ext_vector_type(8)))  _Float16 v8h;
typedef __attribute__((ext_vector_type(8)))  float    v8f;
typedef __attribute__((ext_vector_type(4)))  unsigned int v4u;
typedef __attribute__((ext_vector_type(8)))  int      v8i;
typedef __attribute__((ext_vector_type(4)))  int      v4i;

#define DPCA_B   2
#define DPCA_C   256
#define DPCA_L   4096
#define DPCA_BH  16
#define DPCA_D   32
#define DPCA_TK  256
#define DPCA_NP  (DPCA_BH * DPCA_L)   // 65536 points

#if defined(__gfx1250__) && __has_builtin(__builtin_amdgcn_tensor_load_to_lds) && \
    __has_builtin(__builtin_amdgcn_s_wait_tensorcnt)
#define DPCA_TDM 1
#else
#define DPCA_TDM 0
#endif

// ---------------------------------------------------------------------------
// TDM 2-D tile load: rows x width f16, global row stride `stride` (elements),
// packed contiguously into LDS. D# per CDNA5 ISA ch.8 (group0/group1).
// This toolchain exposes the 6-arg builtin:
//   (uint32x4 g0, int32x8 g1, int32x4 g2, int32x4 g3, int32x8 g4, i32 cpol)
// ---------------------------------------------------------------------------
#if DPCA_TDM
__device__ __forceinline__ void dpca_tdm_load_2d(void* lds_dst, const void* gsrc,
                                                 unsigned width, unsigned rows,
                                                 unsigned stride) {
    unsigned long long ga = (unsigned long long)gsrc;
    unsigned lds = (unsigned)(unsigned long long)lds_dst;   // generic->LDS offset (low 32b)
    v4u g0;
    g0[0] = 1u;                                              // count=1 (valid user D#)
    g0[1] = lds;                                             // lds_addr
    g0[2] = (unsigned)ga;                                    // global_addr[31:0]
    g0[3] = ((unsigned)(ga >> 32) & 0x01FFFFFFu) | 0x80000000u; // addr[56:32] | type=2
    v8i g1;
    g1[0] = 0x00010000;                                      // data_size=1 (2 bytes/elem)
    g1[1] = (int)((width & 0xFFFFu) << 16);                  // tensor_dim0[15:0]
    g1[2] = (int)((width >> 16) | ((rows & 0xFFFFu) << 16)); // tensor_dim0[31:16]|dim1[15:0]
    g1[3] = (int)(((rows >> 16) & 0xFFFFu) | ((width & 0xFFFFu) << 16)); // dim1 hi | tile_dim0
    g1[4] = (int)(rows & 0xFFFFu);                           // tile_dim1 (tile_dim2 = 0)
    g1[5] = (int)stride;                                     // tensor_dim0_stride[31:0]
    g1[6] = 0;                                               // stride hi, dim1_stride lo
    g1[7] = 0;
    v4i gz4 = {0, 0, 0, 0};
    v8i gz8 = {0, 0, 0, 0, 0, 0, 0, 0};
    __builtin_amdgcn_tensor_load_to_lds(g0, g1, gz4, gz4, gz8, 0);
}
#endif

// ---------------------------------------------------------------------------
// WMMA fragment helpers (wave32; A/B k order is contiguous k = 16*half + s)
// ---------------------------------------------------------------------------
// A: 16x32 (MxK) f16, row-major, vectorized (two b128 per lane). Needs 16B align.
__device__ __forceinline__ v16h dpca_load_a_vec(const _Float16* A, int lda) {
    int lane = threadIdx.x & 31;
    int half = lane >> 4, r = lane & 15;
    const _Float16* base = A + (size_t)r * lda + half * 16;
    v8h lo = *(const v8h*)base;
    v8h hi = *(const v8h*)(base + 8);
    v16h a;
#pragma unroll
    for (int j = 0; j < 8; ++j) { a[j] = lo[j]; a[j + 8] = hi[j]; }
    return a;
}

// B: 32x16 (KxN) where memory holds N rows of K contiguous (ld=ldn). Vectorized.
__device__ __forceinline__ v16h dpca_load_b_nmajor_vec(const _Float16* Bp, int ldn) {
    int lane = threadIdx.x & 31;
    int half = lane >> 4, c = lane & 15;
    const _Float16* base = Bp + (size_t)c * ldn + half * 16;
    v8h lo = *(const v8h*)base;
    v8h hi = *(const v8h*)(base + 8);
    v16h b;
#pragma unroll
    for (int j = 0; j < 8; ++j) { b[j] = lo[j]; b[j + 8] = hi[j]; }
    return b;
}

// B: 32x16 (KxN) row-major (K rows of ldb) — scalar, used only on LDS tiles.
__device__ __forceinline__ v16h dpca_load_b_rm_lds(const _Float16* Bp, int ldb) {
    int lane = threadIdx.x & 31;
    int half = lane >> 4, c = lane & 15;
    v16h b;
#pragma unroll
    for (int j = 0; j < 16; ++j)
        b[j] = Bp[(size_t)(half * 16 + j) * ldb + c];
    return b;
}

// C/D 16x16 f32: VGPR r, lanes 0-15 -> M=r, lanes 16-31 -> M=8+r, N=lane&15
__device__ __forceinline__ void dpca_store_c(float* Cp, int ldc, v8f acc) {
    int lane = threadIdx.x & 31;
    int half = lane >> 4, c = lane & 15;
#pragma unroll
    for (int r = 0; r < 8; ++r)
        Cp[(size_t)(half * 8 + r) * ldc + c] = acc[r];
}

#define DPCA_WMMA(a, b, acc) \
    __builtin_amdgcn_wmma_f32_16x16x32_f16(false, (a), false, (b), (short)0, (acc), false, false)

// ---------------------------------------------------------------------------
// Stage 1: channel layer-norm (over C=256 per position) -> f16
// ---------------------------------------------------------------------------
__global__ void dpca_ln_f16(const float* __restrict__ x, const float* __restrict__ gamma,
                            const float* __restrict__ beta, _Float16* __restrict__ xn) {
    int batch = blockIdx.y;
    int pos = blockIdx.x * blockDim.x + threadIdx.x;
    if (pos >= DPCA_L) return;
    const float* xb = x + (size_t)batch * DPCA_C * DPCA_L + pos;
    float s = 0.f, ss = 0.f;
    for (int ch = 0; ch < DPCA_C; ++ch) {
        float v = xb[(size_t)ch * DPCA_L];
        s += v; ss += v * v;
    }
    float mean = s * (1.0f / DPCA_C);
    float var = ss * (1.0f / DPCA_C) - mean * mean;
    float inv = 1.0f / (sqrtf(fmaxf(var, 0.f)) + 1e-6f);
    _Float16* yb = xn + (size_t)batch * DPCA_C * DPCA_L + pos;
    for (int ch = 0; ch < DPCA_C; ++ch) {
        float v = (xb[(size_t)ch * DPCA_L] - mean) * inv;
        yb[(size_t)ch * DPCA_L] = (_Float16)(gamma[ch] * v + beta[ch]);
    }
}

__global__ void dpca_cvt_f16(const float* __restrict__ src, _Float16* __restrict__ dst, int n) {
    int i = blockIdx.x * blockDim.x + threadIdx.x;
    if (i < n) dst[i] = (_Float16)src[i];
}

// ---------------------------------------------------------------------------
// Stage 2: conv1x1 GEMMs. 8-wave blocks, 32(O)x64(pos) output per block,
// X panel (256 x 64 f16, row stride L) staged into LDS by the TDM.
// ---------------------------------------------------------------------------
__global__ void dpca_gemm_q(const _Float16* __restrict__ W, const _Float16* __restrict__ X,
                            float* __restrict__ q) {
    __shared__ __align__(16) _Float16 Xt[DPCA_C * 64];   // 32 KB, [ch][pos]
    int p0 = blockIdx.x * 64, og = blockIdx.y * 32, batch = blockIdx.z;
    const _Float16* Xb = X + (size_t)batch * DPCA_C * DPCA_L;
#if DPCA_TDM
    if (threadIdx.x < 32) {
        dpca_tdm_load_2d(Xt, Xb + p0, 64, DPCA_C, DPCA_L);
        __builtin_amdgcn_s_wait_tensorcnt(0);
    }
#else
    for (int i = threadIdx.x; i < DPCA_C * 16; i += blockDim.x) {   // 16 u64 per 64-f16 row
        int ch = i >> 4, off = i & 15;
        ((unsigned long long*)Xt)[i] =
            ((const unsigned long long*)(Xb + (size_t)ch * DPCA_L + p0))[off];
    }
#endif
    __syncthreads();
    int wave = threadIdx.x >> 5;
    int ot = og + (wave >> 2) * 16;     // output-row tile
    int ptl = (wave & 3) * 16;          // local position tile
    v8f acc = {};
    for (int kk = 0; kk < DPCA_C; kk += 32) {
        v16h a = dpca_load_a_vec(W + (size_t)ot * DPCA_C + kk, DPCA_C);
        v16h b = dpca_load_b_rm_lds(Xt + (size_t)kk * 64 + ptl, 64);
        acc = DPCA_WMMA(a, b, acc);
    }
    int lane = threadIdx.x & 31, half = lane >> 4, col = lane & 15;
#pragma unroll
    for (int r = 0; r < 8; ++r) {
        int o = ot + half * 8 + r;
        int pos = p0 + ptl + col;
        int bh = batch * 8 + (o >> 5);
        q[((size_t)bh * DPCA_L + pos) * DPCA_D + (o & 31)] = acc[r];
    }
}

__global__ void dpca_gemm_kv(const _Float16* __restrict__ W, const _Float16* __restrict__ X,
                             float* __restrict__ karr, float* __restrict__ varr) {
    __shared__ __align__(16) _Float16 Xt[DPCA_C * 64];
    int p0 = blockIdx.x * 64, og = blockIdx.y * 32, batch = blockIdx.z;  // og in [0,512)
    const _Float16* Xb = X + (size_t)batch * DPCA_C * DPCA_L;
#if DPCA_TDM
    if (threadIdx.x < 32) {
        dpca_tdm_load_2d(Xt, Xb + p0, 64, DPCA_C, DPCA_L);
        __builtin_amdgcn_s_wait_tensorcnt(0);
    }
#else
    for (int i = threadIdx.x; i < DPCA_C * 16; i += blockDim.x) {
        int ch = i >> 4, off = i & 15;
        ((unsigned long long*)Xt)[i] =
            ((const unsigned long long*)(Xb + (size_t)ch * DPCA_L + p0))[off];
    }
#endif
    __syncthreads();
    int wave = threadIdx.x >> 5;
    int ot = og + (wave >> 2) * 16;
    int ptl = (wave & 3) * 16;
    v8f acc = {};
    for (int kk = 0; kk < DPCA_C; kk += 32) {
        v16h a = dpca_load_a_vec(W + (size_t)ot * DPCA_C + kk, DPCA_C);
        v16h b = dpca_load_b_rm_lds(Xt + (size_t)kk * 64 + ptl, 64);
        acc = DPCA_WMMA(a, b, acc);
    }
    int lane = threadIdx.x & 31, half = lane >> 4, col = lane & 15;
#pragma unroll
    for (int r = 0; r < 8; ++r) {
        int o = ot + half * 8 + r;
        int pos = p0 + ptl + col;
        float* dst = (o < 256) ? karr : varr;
        int oo = o & 255;
        int bh = batch * 8 + (oo >> 5);
        dst[((size_t)bh * DPCA_L + pos) * DPCA_D + (oo & 31)] = acc[r];
    }
}

// ---------------------------------------------------------------------------
// Stage 3: per-position L2 norm over 32 dims, in-place + f16 copy
// ---------------------------------------------------------------------------
__global__ void dpca_l2norm(float* __restrict__ t, _Float16* __restrict__ th, int nrows) {
    int i = blockIdx.x * blockDim.x + threadIdx.x;
    if (i >= nrows) return;
    float* r = t + (size_t)i * DPCA_D;
    float ss = 0.f;
#pragma unroll
    for (int d = 0; d < DPCA_D; ++d) ss += r[d] * r[d];
    float inv = 1.0f / fmaxf(sqrtf(ss), 1e-12f);
    _Float16* rh = th + (size_t)i * DPCA_D;
#pragma unroll
    for (int d = 0; d < DPCA_D; ++d) {
        float v = r[d] * inv;
        r[d] = v; rh[d] = (_Float16)v;
    }
}

// ---------------------------------------------------------------------------
// Stage 4: k-means (k=256, d=32) on 65536 q points
// ---------------------------------------------------------------------------
__global__ void dpca_km_init(const float* __restrict__ qf, float* __restrict__ cent,
                             float* __restrict__ sums, float* __restrict__ cnts) {
    int i = blockIdx.x * blockDim.x + threadIdx.x;   // 8192
    if (i >= DPCA_TK * DPCA_D) return;
    cent[i] = qf[i];         // q_flat[:256] == bh0, pos 0..255
    sums[i] = 0.f;
    if (i < DPCA_TK) cnts[i] = 0.f;
}

__global__ void dpca_cent_finalize(const float* __restrict__ cent, _Float16* __restrict__ centh,
                                   float* __restrict__ cnorm, float* cnts) {
    int j = blockIdx.x * blockDim.x + threadIdx.x;
    if (j >= DPCA_TK) return;
    float ss = 0.f;
#pragma unroll
    for (int d = 0; d < DPCA_D; ++d) {
        float v = cent[j * DPCA_D + d];
        ss += v * v;
        centh[j * DPCA_D + d] = (_Float16)v;
    }
    cnorm[j] = ss;
    if (cnts) cnts[j] = 0.f;
}

// score panel: 32 points x 256 centroids via 32 WMMAs (2 waves), into LDS
__device__ __forceinline__ void dpca_score_panel(const _Float16* __restrict__ Xh,
                                                 const _Float16* __restrict__ CH,
                                                 float* __restrict__ S) {
    int wave = threadIdx.x >> 5;
    v16h a = dpca_load_a_vec(Xh + (size_t)(wave * 16) * DPCA_D, DPCA_D);
#pragma unroll 4
    for (int nt = 0; nt < 16; ++nt) {
        v16h b = dpca_load_b_nmajor_vec(CH + (size_t)nt * 16 * DPCA_D, DPCA_D);
        v8f acc = {};
        acc = DPCA_WMMA(a, b, acc);
        dpca_store_c(S + (size_t)(wave * 16) * DPCA_TK + nt * 16, DPCA_TK, acc);
    }
}

#if DPCA_TDM
#define DPCA_STAGE_CENTROIDS(CH, centh)                                              \
    if (threadIdx.x < 32) {                                                          \
        dpca_tdm_load_2d((CH), (centh), DPCA_TK * DPCA_D, 1, DPCA_TK * DPCA_D);      \
        __builtin_amdgcn_s_wait_tensorcnt(0);                                        \
    }                                                                                \
    __syncthreads();
#else
#define DPCA_STAGE_CENTROIDS(CH, centh)                                              \
    for (int i = threadIdx.x; i < DPCA_TK * DPCA_D / 4; i += blockDim.x)             \
        ((unsigned long long*)(CH))[i] = ((const unsigned long long*)(centh))[i];    \
    __syncthreads();
#endif

__global__ void dpca_km_assign(const _Float16* __restrict__ qh, const float* __restrict__ qf,
                               const _Float16* __restrict__ centh, const float* __restrict__ cnorm,
                               float* __restrict__ sums, float* __restrict__ cnts) {
    __shared__ __align__(16) _Float16 CH[DPCA_TK * DPCA_D];   // 16 KB
    __shared__ __align__(16) float S[32 * DPCA_TK];           // 32 KB
    DPCA_STAGE_CENTROIDS(CH, centh)
    int base = blockIdx.x * 32;
    dpca_score_panel(qh + (size_t)base * DPCA_D, CH, S);
    __syncthreads();
    int t = threadIdx.x;
    if (t < 32) {
        const float* row = S + (size_t)t * DPCA_TK;
        float best = cnorm[0] - 2.f * row[0];
        int bj = 0;
        for (int j = 1; j < DPCA_TK; ++j) {
            float d = cnorm[j] - 2.f * row[j];
            if (d < best) { best = d; bj = j; }
        }
        const float* x = qf + (size_t)(base + t) * DPCA_D;
#pragma unroll
        for (int d = 0; d < DPCA_D; ++d) atomicAdd(&sums[bj * DPCA_D + d], x[d]);
        atomicAdd(&cnts[bj], 1.0f);
    }
}

__global__ void dpca_km_update(float* __restrict__ cent, float* __restrict__ sums,
                               const float* __restrict__ cnts) {
    int i = blockIdx.x * blockDim.x + threadIdx.x;   // 8192
    if (i >= DPCA_TK * DPCA_D) return;
    float c = cnts[i >> 5];
    if (c > 0.f) cent[i] = sums[i] / fmaxf(c, 1.0f);
    sums[i] = 0.f;
}

// assign each k point, L1 distance to its centroid
__global__ void dpca_k_dist(const _Float16* __restrict__ kh, const float* __restrict__ kf,
                            const _Float16* __restrict__ centh, const float* __restrict__ cnorm,
                            const float* __restrict__ centf, float* __restrict__ kdist) {
    __shared__ __align__(16) _Float16 CH[DPCA_TK * DPCA_D];
    __shared__ __align__(16) float S[32 * DPCA_TK];
    DPCA_STAGE_CENTROIDS(CH, centh)
    int base = blockIdx.x * 32;
    dpca_score_panel(kh + (size_t)base * DPCA_D, CH, S);
    __syncthreads();
    int t = threadIdx.x;
    if (t < 32) {
        const float* row = S + (size_t)t * DPCA_TK;
        float best = cnorm[0] - 2.f * row[0];
        int bj = 0;
        for (int j = 1; j < DPCA_TK; ++j) {
            float d = cnorm[j] - 2.f * row[j];
            if (d < best) { best = d; bj = j; }
        }
        const float* x = kf + (size_t)(base + t) * DPCA_D;
        const float* c = centf + (size_t)bj * DPCA_D;
        float l1 = 0.f;
#pragma unroll
        for (int d = 0; d < DPCA_D; ++d) l1 += fabsf(c[d] - x[d]);
        kdist[base + t] = l1;
    }
}

// ---------------------------------------------------------------------------
// Stage 5: per-batch-head top-256 of 4096 (largest; lower index on ties)
// ---------------------------------------------------------------------------
__global__ void dpca_topk(const float* __restrict__ kdist, int* __restrict__ topk) {
    __shared__ float vals[DPCA_L];   // 16 KB
    __shared__ float rv[256];
    __shared__ int   ri[256];
    int bh = blockIdx.x;
    const float* src = kdist + (size_t)bh * DPCA_L;
    for (int i = threadIdx.x; i < DPCA_L; i += blockDim.x) vals[i] = src[i];
    __syncthreads();
    for (int t = 0; t < DPCA_TK; ++t) {
        float bv = -__builtin_inff();
        int bi = 0x7fffffff;
        for (int i = threadIdx.x; i < DPCA_L; i += blockDim.x) {
            float v = vals[i];
            if (v > bv || (v == bv && i < bi)) { bv = v; bi = i; }
        }
        rv[threadIdx.x] = bv; ri[threadIdx.x] = bi;
        __syncthreads();
        for (int s = 128; s > 0; s >>= 1) {
            if (threadIdx.x < (unsigned)s) {
                float ov = rv[threadIdx.x + s]; int oi = ri[threadIdx.x + s];
                if (ov > rv[threadIdx.x] || (ov == rv[threadIdx.x] && oi < ri[threadIdx.x])) {
                    rv[threadIdx.x] = ov; ri[threadIdx.x] = oi;
                }
            }
            __syncthreads();
        }
        if (threadIdx.x == 0) {
            topk[bh * DPCA_TK + t] = ri[0];
            vals[ri[0]] = -__builtin_inff();
        }
        __syncthreads();
    }
}

// gather: k kept row-major [bh][key][32]; v stored TRANSPOSED [bh][32][256]
__global__ void dpca_gather(const float* __restrict__ kf, const float* __restrict__ vf,
                            const int* __restrict__ topk,
                            _Float16* __restrict__ kselh, _Float16* __restrict__ vselT) {
    int idx = blockIdx.x * blockDim.x + threadIdx.x;   // bh*TK + t
    if (idx >= DPCA_BH * DPCA_TK) return;
    int bh = idx >> 8, t = idx & 255;
    int src = topk[idx];
    const float* kr = kf + ((size_t)bh * DPCA_L + src) * DPCA_D;
    const float* vr = vf + ((size_t)bh * DPCA_L + src) * DPCA_D;
    _Float16* ko = kselh + (size_t)idx * DPCA_D;
#pragma unroll
    for (int d = 0; d < DPCA_D; ++d) {
        ko[d] = (_Float16)kr[d];
        vselT[((size_t)bh * DPCA_D + d) * DPCA_TK + t] = (_Float16)vr[d];
    }
}

// ---------------------------------------------------------------------------
// Stage 6: attention: sim(32x256) -> softmax -> out(32x32), per 2-wave block
// ---------------------------------------------------------------------------
__global__ void dpca_attn(const _Float16* __restrict__ qh, const _Float16* __restrict__ kselh,
                          const _Float16* __restrict__ vselT, _Float16* __restrict__ outh) {
    __shared__ __align__(16) float S[32 * DPCA_TK];      // 32 KB scores
    __shared__ __align__(16) _Float16 P[32 * DPCA_TK];   // 16 KB probs
    int bh = blockIdx.y;
    int base = blockIdx.x * 32;
    int wave = threadIdx.x >> 5;
    v16h a = dpca_load_a_vec(qh + ((size_t)bh * DPCA_L + base + wave * 16) * DPCA_D, DPCA_D);
    const _Float16* Kb = kselh + (size_t)bh * DPCA_TK * DPCA_D;
#pragma unroll 4
    for (int nt = 0; nt < 16; ++nt) {
        __builtin_prefetch(Kb + (size_t)(nt + 1) * 16 * DPCA_D, 0, 3);
        v16h b = dpca_load_b_nmajor_vec(Kb + (size_t)nt * 16 * DPCA_D, DPCA_D);
        v8f acc = {};
        acc = DPCA_WMMA(a, b, acc);
        dpca_store_c(S + (size_t)(wave * 16) * DPCA_TK + nt * 16, DPCA_TK, acc);
    }
    __syncthreads();
    int t = threadIdx.x;
    if (t < 32) {
        float* row = S + (size_t)t * DPCA_TK;
        float m = row[0];
        for (int j = 1; j < DPCA_TK; ++j) m = fmaxf(m, row[j]);
        float s = 0.f;
        for (int j = 0; j < DPCA_TK; ++j) {
            float e = __expf(row[j] - m);
            s += e; row[j] = e;
        }
        float inv = 1.0f / s;
        _Float16* ph = P + (size_t)t * DPCA_TK;
        for (int j = 0; j < DPCA_TK; ++j) ph[j] = (_Float16)(row[j] * inv);
    }
    __syncthreads();
    const _Float16* Vb = vselT + (size_t)bh * DPCA_D * DPCA_TK;
    int lane = threadIdx.x & 31, half = lane >> 4, col = lane & 15;
#pragma unroll
    for (int nt = 0; nt < 2; ++nt) {
        v8f acc = {};
        for (int kk = 0; kk < DPCA_TK; kk += 32) {
            v16h pa = dpca_load_a_vec(P + (size_t)(wave * 16) * DPCA_TK + kk, DPCA_TK);
            // V is [bh][d][256]: n-major rows of K-contiguous -> vector loads
            v16h vb = dpca_load_b_nmajor_vec(Vb + (size_t)(nt * 16) * DPCA_TK + kk, DPCA_TK);
            acc = DPCA_WMMA(pa, vb, acc);
        }
#pragma unroll
        for (int r = 0; r < 8; ++r) {
            int pos = base + wave * 16 + half * 8 + r;
            int d = nt * 16 + col;
            outh[((size_t)bh * DPCA_L + pos) * DPCA_D + d] = (_Float16)acc[r];
        }
    }
}

// ---------------------------------------------------------------------------
// Stage 7: unfold + output conv1x1 (W_out 256x256) -> Y[b][256][L]
// The unfold panel for 64 positions is 8 contiguous 4 KB runs (one per head-
// part) with stride L*32 -> one 2-D TDM tile into LDS [hp][pos][32].
// ---------------------------------------------------------------------------
__global__ void dpca_gemm_out(const _Float16* __restrict__ W, const _Float16* __restrict__ outh,
                              float* __restrict__ Y) {
    __shared__ __align__(16) _Float16 Xt[8 * 64 * DPCA_D];   // 32 KB
    int p0 = blockIdx.x * 64, og = blockIdx.y * 32, batch = blockIdx.z;
    const _Float16* src0 = outh + ((size_t)(batch * 8) * DPCA_L + p0) * DPCA_D;
#if DPCA_TDM
    if (threadIdx.x < 32) {
        dpca_tdm_load_2d(Xt, src0, 64 * DPCA_D, 8, DPCA_L * DPCA_D);
        __builtin_amdgcn_s_wait_tensorcnt(0);
    }
#else
    for (int i = threadIdx.x; i < 8 * 512; i += blockDim.x) {   // 512 u64 per head-part
        int hp = i >> 9, off = i & 511;
        ((unsigned long long*)Xt)[i] =
            ((const unsigned long long*)(src0 + (size_t)hp * DPCA_L * DPCA_D))[off];
    }
#endif
    __syncthreads();
    int wave = threadIdx.x >> 5;
    int ot = og + (wave >> 2) * 16;
    int ptl = (wave & 3) * 16;
    int lane = threadIdx.x & 31, half = lane >> 4, col = lane & 15;
    v8f acc = {};
    for (int kk = 0; kk < DPCA_C; kk += 32) {
        v16h a = dpca_load_a_vec(W + (size_t)ot * DPCA_C + kk, DPCA_C);
        v16h b;
#pragma unroll
        for (int j = 0; j < 16; ++j) {
            int k = kk + half * 16 + j;   // inner channel
            b[j] = Xt[((size_t)(k >> 5) * 64 + ptl + col) * DPCA_D + (k & 31)];
        }
        acc = DPCA_WMMA(a, b, acc);
    }
#pragma unroll
    for (int r = 0; r < 8; ++r) {
        int o = ot + half * 8 + r;
        Y[((size_t)batch * DPCA_C + o) * DPCA_L + p0 + ptl + col] = acc[r];
    }
}

// Stage 8: final channel-LN + gamma_scale * out + residual
__global__ void dpca_ln_out(const float* __restrict__ Y, const float* __restrict__ gamma,
                            const float* __restrict__ beta, const float* __restrict__ gscale,
                            const float* __restrict__ qsrc, float* __restrict__ out) {
    int batch = blockIdx.y;
    int pos = blockIdx.x * blockDim.x + threadIdx.x;
    if (pos >= DPCA_L) return;
    const float* yb = Y + (size_t)batch * DPCA_C * DPCA_L + pos;
    float s = 0.f, ss = 0.f;
    for (int ch = 0; ch < DPCA_C; ++ch) {
        float v = yb[(size_t)ch * DPCA_L];
        s += v; ss += v * v;
    }
    float mean = s * (1.0f / DPCA_C);
    float var = ss * (1.0f / DPCA_C) - mean * mean;
    float inv = 1.0f / (sqrtf(fmaxf(var, 0.f)) + 1e-6f);
    float g = gscale[0];
    const float* qb = qsrc + (size_t)batch * DPCA_C * DPCA_L + pos;
    float* ob = out + (size_t)batch * DPCA_C * DPCA_L + pos;
    for (int ch = 0; ch < DPCA_C; ++ch) {
        float v = gamma[ch] * ((yb[(size_t)ch * DPCA_L] - mean) * inv) + beta[ch];
        ob[(size_t)ch * DPCA_L] = g * v + qb[(size_t)ch * DPCA_L];
    }
}

// ---------------------------------------------------------------------------
extern "C" void kernel_launch(void* const* d_in, const int* in_sizes, int n_in,
                              void* d_out, int out_size, void* d_ws, size_t ws_size,
                              hipStream_t stream) {
    const float* qsrc  = (const float*)d_in[0];
    const float* ctx   = (const float*)d_in[1];
    const float* w_q   = (const float*)d_in[2];
    const float* w_kv  = (const float*)d_in[3];
    const float* w_out = (const float*)d_in[4];
    const float* g_ctx = (const float*)d_in[5];
    const float* b_ctx = (const float*)d_in[6];
    const float* g_q   = (const float*)d_in[7];
    const float* b_q   = (const float*)d_in[8];
    const float* g_o   = (const float*)d_in[9];
    const float* b_o   = (const float*)d_in[10];
    const float* gsc   = (const float*)d_in[11];
    float* out = (float*)d_out;

    char* wp = (char*)d_ws;
    auto alloc = [&](size_t bytes) -> void* {
        void* p = (void*)wp;
        wp += (bytes + 255) & ~(size_t)255;
        return p;
    };
    const size_t NE = (size_t)DPCA_B * DPCA_C * DPCA_L;
    const size_t NQ = (size_t)DPCA_BH * DPCA_L * DPCA_D;

    _Float16* ctxn  = (_Float16*)alloc(NE * 2);
    _Float16* qsn   = (_Float16*)alloc(NE * 2);
    _Float16* wqh   = (_Float16*)alloc((size_t)256 * 256 * 2);
    _Float16* wkvh  = (_Float16*)alloc((size_t)512 * 256 * 2);
    _Float16* wouth = (_Float16*)alloc((size_t)256 * 256 * 2);
    float*    qf    = (float*)alloc(NQ * 4);
    _Float16* qh    = (_Float16*)alloc(NQ * 2);
    float*    kf    = (float*)alloc(NQ * 4);
    _Float16* kh    = (_Float16*)alloc(NQ * 2);
    float*    vf    = (float*)alloc(NQ * 4);
    float*    cent  = (float*)alloc((size_t)DPCA_TK * DPCA_D * 4);
    _Float16* centh = (_Float16*)alloc((size_t)DPCA_TK * DPCA_D * 2);
    float*    cnorm = (float*)alloc((size_t)DPCA_TK * 4);
    float*    sums  = (float*)alloc((size_t)DPCA_TK * DPCA_D * 4);
    float*    cnts  = (float*)alloc((size_t)DPCA_TK * 4);
    float*    kdist = (float*)alloc((size_t)DPCA_NP * 4);
    int*      tkid  = (int*)alloc((size_t)DPCA_BH * DPCA_TK * 4);
    _Float16* kselh = (_Float16*)alloc((size_t)DPCA_BH * DPCA_TK * DPCA_D * 2);
    _Float16* vselT = (_Float16*)alloc((size_t)DPCA_BH * DPCA_D * DPCA_TK * 2);
    _Float16* outh  = (_Float16*)alloc(NQ * 2);
    float*    Y     = (float*)alloc(NE * 4);

    // 1) layer-norms -> f16, weight conversions
    dpca_ln_f16<<<dim3(DPCA_L / 256, DPCA_B), 256, 0, stream>>>(ctx, g_ctx, b_ctx, ctxn);
    dpca_ln_f16<<<dim3(DPCA_L / 256, DPCA_B), 256, 0, stream>>>(qsrc, g_q, b_q, qsn);
    dpca_cvt_f16<<<(256 * 256 + 255) / 256, 256, 0, stream>>>(w_q, wqh, 256 * 256);
    dpca_cvt_f16<<<(512 * 256 + 255) / 256, 256, 0, stream>>>(w_kv, wkvh, 512 * 256);
    dpca_cvt_f16<<<(256 * 256 + 255) / 256, 256, 0, stream>>>(w_out, wouth, 256 * 256);

    // 2) conv1x1 projections (TDM-staged WMMA)
    dpca_gemm_q<<<dim3(DPCA_L / 64, 256 / 32, DPCA_B), 256, 0, stream>>>(wqh, qsn, qf);
    dpca_gemm_kv<<<dim3(DPCA_L / 64, 512 / 32, DPCA_B), 256, 0, stream>>>(wkvh, ctxn, kf, vf);

    // 3) L2 normalize q and k, produce f16 copies
    dpca_l2norm<<<(DPCA_NP + 255) / 256, 256, 0, stream>>>(qf, qh, DPCA_NP);
    dpca_l2norm<<<(DPCA_NP + 255) / 256, 256, 0, stream>>>(kf, kh, DPCA_NP);

    // 4) k-means fit on q points (10 iters, WMMA scoring + L2 atomics)
    dpca_km_init<<<(DPCA_TK * DPCA_D + 255) / 256, 256, 0, stream>>>(qf, cent, sums, cnts);
    dpca_cent_finalize<<<1, 256, 0, stream>>>(cent, centh, cnorm, (float*)nullptr);
    for (int it = 0; it < 10; ++it) {
        dpca_km_assign<<<DPCA_NP / 32, 64, 0, stream>>>(qh, qf, centh, cnorm, sums, cnts);
        dpca_km_update<<<(DPCA_TK * DPCA_D + 255) / 256, 256, 0, stream>>>(cent, sums, cnts);
        dpca_cent_finalize<<<1, 256, 0, stream>>>(cent, centh, cnorm, cnts);
    }

    // 5) assign k points, L1 dist, top-256 per batch-head, gather (v transposed)
    dpca_k_dist<<<DPCA_NP / 32, 64, 0, stream>>>(kh, kf, centh, cnorm, cent, kdist);
    dpca_topk<<<DPCA_BH, 256, 0, stream>>>(kdist, tkid);
    dpca_gather<<<(DPCA_BH * DPCA_TK + 255) / 256, 256, 0, stream>>>(kf, vf, tkid, kselh, vselT);

    // 6) attention (WMMA QK^T -> softmax -> WMMA PV)
    dpca_attn<<<dim3(DPCA_L / 32, DPCA_BH), 64, 0, stream>>>(qh, kselh, vselT, outh);

    // 7) output projection (TDM-staged unfold) + final LN + residual
    dpca_gemm_out<<<dim3(DPCA_L / 64, 256 / 32, DPCA_B), 256, 0, stream>>>(wouth, outh, Y);
    dpca_ln_out<<<dim3(DPCA_L / 256, DPCA_B), 256, 0, stream>>>(Y, g_o, b_o, gsc, qsrc, out);
}
